// Matcher_55594056680092
// MI455X (gfx1250) — compile-verified, matrix operand order checked by reading, not run on hardware
//
#include <hip/hip_runtime.h>
#include <cstdint>
#include <cstddef>

typedef __attribute__((ext_vector_type(16))) _Float16 v16h;
typedef __attribute__((ext_vector_type(8)))  _Float16 v8h;
typedef __attribute__((ext_vector_type(8)))  float    v8f;

#define LDSK 40   // 32 K halves + 8 pad -> 80B row stride (16B aligned, no pow2 bank walk)

// Build a 16-bit A/B WMMA fragment from an LDS tile row.
// Layout (CDNA5 ISA): lane L holds row = L%16; with h = L/16 the lane's
// 16 K-values are the 8 contiguous halves at K = h*8 and the 8 at K = 16 + h*8.
__device__ __forceinline__ v16h load_frag(const _Float16* rowp, int h) {
  v8h lo = *(const v8h*)(rowp + h * 8);
  v8h hi = *(const v8h*)(rowp + 16 + h * 8);
  return __builtin_shufflevector(lo, hi, 0,1,2,3,4,5,6,7,8,9,10,11,12,13,14,15);
}

// ---------------------------------------------------------------------------
// Row inverse L2 norm: one 256-thread block per 1024-element row.
// ---------------------------------------------------------------------------
__global__ __launch_bounds__(256)
void rownorm_kernel(const float* __restrict__ X, float* __restrict__ inv) {
  __shared__ float red[256];
  const int t = threadIdx.x;
  const float* p = X + (size_t)blockIdx.x * 1024;
  float s = 0.f;
#pragma unroll
  for (int i = 0; i < 4; ++i) { float v = p[t + i * 256]; s += v * v; }
  red[t] = s; __syncthreads();
  for (int off = 128; off > 0; off >>= 1) {
    if (t < off) red[t] += red[t + off];
    __syncthreads();
  }
  if (t == 0) inv[blockIdx.x] = rsqrtf(red[0]);
}

// ---------------------------------------------------------------------------
// Fourier positional embedding, written TRANSPOSED as f16: posT[d][k], d<512 -> sin,
// d>=512 -> cos, k indexes the 48x48 grid (x,y in [-1,1]).
// ---------------------------------------------------------------------------
__global__ __launch_bounds__(256)
void pos_emb_kernel(const float* __restrict__ omega, const float* __restrict__ scale_p,
                    _Float16* __restrict__ posT) {
  const int idx = blockIdx.x * 256 + threadIdx.x;   // d*2304 + k
  if (idx >= 1024 * 2304) return;
  const int d = idx / 2304;
  const int k = idx - d * 2304;
  const int j = d & 511;
  const int iy = k / 48;
  const int ix = k - iy * 48;
  const float gx = -1.f + (2.f / 47.f) * (float)ix;
  const float gy = -1.f + (2.f / 47.f) * (float)iy;
  const float e = scale_p[0] * (omega[2 * j] * gx + omega[2 * j + 1] * gy);
  const float v = (d < 512) ? __sinf(e) : __cosf(e);
  posT[idx] = (_Float16)v;
}

// ---------------------------------------------------------------------------
// Batched NT GEMM via v_wmma_f32_16x16x32_f16.
//   C[m][n] = alpha * sum_k a(m,k) * b(n,k)
// A source: f32 row-major [M][K] (optionally scaled per-row by invA).
// B source: either f32 row-major [N][K] scaled by invB, or f16 row-major [N][K].
// Block: 128 threads = 4 wave32 waves, block tile 128x128, wave tile 64x64,
// K staged through LDS in steps of 32 with on-the-fly f32->f16 conversion.
// ---------------------------------------------------------------------------
template <bool A_SCALED, bool B_HALF>
__global__ __launch_bounds__(128)
void gemm_nt_wmma(const float* __restrict__ A,
                  const float* __restrict__ Bf,
                  const _Float16* __restrict__ Bh,
                  const float* __restrict__ invA,
                  const float* __restrict__ invB,
                  float* __restrict__ C,
                  int M, int N, int K,
                  long strideA, long strideB, long strideC, long strideInv,
                  float alpha) {
  const int z = blockIdx.z;
  A += (size_t)z * strideA;
  if (B_HALF) Bh += (size_t)z * strideB; else Bf += (size_t)z * strideB;
  C += (size_t)z * strideC;
  if (A_SCALED) { invA += (size_t)z * strideInv; invB += (size_t)z * strideInv; }

  __shared__ __align__(16) _Float16 As[128 * LDSK];
  __shared__ __align__(16) _Float16 Bs[128 * LDSK];

  const int t      = threadIdx.x;
  const int wave   = t >> 5;
  const int lane   = t & 31;
  const int lane16 = lane & 15;
  const int h      = lane >> 4;
  const int wm     = wave >> 1;       // wave row   (0..1)
  const int wn     = wave & 1;        // wave col   (0..1)

  const int rowA = blockIdx.y * 128 + t;   // one A-tile row per thread
  const int rowB = blockIdx.x * 128 + t;   // one B-tile row per thread
  const float scaleA = A_SCALED ? invA[rowA] : 1.0f;
  const float scaleB = B_HALF ? 1.0f : invB[rowB];

  v8f acc[4][4] = {};

  for (int kt = 0; kt < K; kt += 32) {
    __syncthreads();   // protect LDS from previous iteration's readers

    // ---- stage A tile: 128 rows x 32 k, f32 -> f16 with per-row scale ----
    {
      const float4* src = (const float4*)(A + (size_t)rowA * K + kt);
      __align__(16) _Float16 tmp[32];
#pragma unroll
      for (int j = 0; j < 8; ++j) {
        float4 v = src[j];
        tmp[4 * j + 0] = (_Float16)(v.x * scaleA);
        tmp[4 * j + 1] = (_Float16)(v.y * scaleA);
        tmp[4 * j + 2] = (_Float16)(v.z * scaleA);
        tmp[4 * j + 3] = (_Float16)(v.w * scaleA);
      }
#pragma unroll
      for (int j = 0; j < 4; ++j)
        *(v8h*)&As[t * LDSK + j * 8] = *(const v8h*)&tmp[j * 8];
      if (kt + 32 < K)
        __builtin_prefetch(A + (size_t)rowA * K + kt + 32, 0, 0);
    }

    // ---- stage B tile ----
    if (B_HALF) {
      const uint4* src = (const uint4*)(Bh + (size_t)rowB * K + kt);
#pragma unroll
      for (int j = 0; j < 4; ++j)
        *(uint4*)&Bs[t * LDSK + j * 8] = src[j];
      if (kt + 32 < K)
        __builtin_prefetch(Bh + (size_t)rowB * K + kt + 32, 0, 0);
    } else {
      const float4* src = (const float4*)(Bf + (size_t)rowB * K + kt);
      __align__(16) _Float16 tmp[32];
#pragma unroll
      for (int j = 0; j < 8; ++j) {
        float4 v = src[j];
        tmp[4 * j + 0] = (_Float16)(v.x * scaleB);
        tmp[4 * j + 1] = (_Float16)(v.y * scaleB);
        tmp[4 * j + 2] = (_Float16)(v.z * scaleB);
        tmp[4 * j + 3] = (_Float16)(v.w * scaleB);
      }
#pragma unroll
      for (int j = 0; j < 4; ++j)
        *(v8h*)&Bs[t * LDSK + j * 8] = *(const v8h*)&tmp[j * 8];
      if (kt + 32 < K)
        __builtin_prefetch(Bf + (size_t)rowB * K + kt + 32, 0, 0);
    }

    __syncthreads();

    // ---- 64x64 wave tile: 4x4 WMMA subtiles over this 32-deep K slice ----
    v16h af[4], bf16[4];
#pragma unroll
    for (int m = 0; m < 4; ++m)
      af[m] = load_frag(&As[(wm * 64 + m * 16 + lane16) * LDSK], h);
#pragma unroll
    for (int n = 0; n < 4; ++n)
      bf16[n] = load_frag(&Bs[(wn * 64 + n * 16 + lane16) * LDSK], h);

#pragma unroll
    for (int m = 0; m < 4; ++m)
#pragma unroll
      for (int n = 0; n < 4; ++n)
        acc[m][n] = __builtin_amdgcn_wmma_f32_16x16x32_f16(
            false, af[m], false, bf16[n], (short)0, acc[m][n], false, false);
  }

  // ---- epilogue: C/D layout -> lane L writes col = L%16, rows v + 8*(L/16) ----
  const int baseRow = blockIdx.y * 128 + wm * 64;
  const int baseCol = blockIdx.x * 128 + wn * 64;
#pragma unroll
  for (int m = 0; m < 4; ++m) {
#pragma unroll
    for (int n = 0; n < 4; ++n) {
      const int c = baseCol + n * 16 + lane16;
#pragma unroll
      for (int v = 0; v < 8; ++v) {
        const int r = baseRow + m * 16 + h * 8 + v;
        C[(size_t)r * N + c] = alpha * acc[m][n][v];
      }
    }
  }
}

// ---------------------------------------------------------------------------
// Row softmax over 2304 logits: one 256-thread block per (b, q) row.
// ---------------------------------------------------------------------------
__global__ __launch_bounds__(256)
void softmax_kernel(const float* __restrict__ logits, float* __restrict__ attn) {
  __shared__ float red[256];
  const size_t row = blockIdx.x;
  const float* p = logits + row * 2304;
  float* o = attn + row * 2304;
  const int t = threadIdx.x;

  float v[9];
  float m = -INFINITY;
#pragma unroll
  for (int i = 0; i < 9; ++i) { v[i] = p[t + i * 256]; m = fmaxf(m, v[i]); }
  red[t] = m; __syncthreads();
  for (int off = 128; off > 0; off >>= 1) {
    if (t < off) red[t] = fmaxf(red[t], red[t + off]);
    __syncthreads();
  }
  m = red[0]; __syncthreads();

  float s = 0.f;
#pragma unroll
  for (int i = 0; i < 9; ++i) { v[i] = __expf(v[i] - m); s += v[i]; }
  red[t] = s; __syncthreads();
  for (int off = 128; off > 0; off >>= 1) {
    if (t < off) red[t] += red[t + off];
    __syncthreads();
  }
  const float inv = 1.0f / red[0];
#pragma unroll
  for (int i = 0; i < 9; ++i) o[t + i * 256] = v[i] * inv;
}

// ---------------------------------------------------------------------------
extern "C" void kernel_launch(void* const* d_in, const int* in_sizes, int n_in,
                              void* d_out, int out_size, void* d_ws, size_t ws_size,
                              hipStream_t stream) {
  (void)in_sizes; (void)n_in; (void)out_size; (void)ws_size;

  const float* fA    = (const float*)d_in[0];   // [8,48,48,1024]
  const float* fB    = (const float*)d_in[1];   // [8,48,48,1024]
  const float* omega = (const float*)d_in[2];   // [512,2]
  const float* scale = (const float*)d_in[3];   // scalar

  const long Bn = 8, HW = 2304, D = 1024;
  float* out    = (float*)d_out;
  float* logits = out;                         // 8*2304*2304
  float* attn   = out + (size_t)Bn * HW * HW;  // 8*2304*2304
  float* match  = attn + (size_t)Bn * HW * HW; // 8*2304*1024

  char* ws = (char*)d_ws;
  float*    invA = (float*)ws;                                   // 18432 f32
  float*    invB = invA + Bn * HW;                               // 18432 f32
  _Float16* posT = (_Float16*)(ws + 2 * (size_t)Bn * HW * 4);    // [1024][2304] f16

  // 1) per-row inverse norms
  rownorm_kernel<<<(int)(Bn * HW), 256, 0, stream>>>(fA, invA);
  rownorm_kernel<<<(int)(Bn * HW), 256, 0, stream>>>(fB, invB);

  // 2) transposed f16 positional embedding table
  pos_emb_kernel<<<(1024 * 2304) / 256, 256, 0, stream>>>(omega, scale, posT);

  // 3) logits = 10 * faN @ fbN^T   (per batch: 2304 x 2304 x 1024)
  {
    dim3 g(18, 18, (unsigned)Bn);
    gemm_nt_wmma<true, false><<<g, 128, 0, stream>>>(
        fA, fB, (const _Float16*)nullptr, invA, invB, logits,
        (int)HW, (int)HW, (int)D,
        HW * D, HW * D, HW * HW, HW, 10.0f);
  }

  // 4) softmax over k
  softmax_kernel<<<(int)(Bn * HW), 256, 0, stream>>>(logits, attn);

  // 5) match = attn @ pos_emb  (per batch: 2304 x 1024 x 2304), B = posT (shared)
  {
    dim3 g(8, 18, (unsigned)Bn);
    gemm_nt_wmma<false, true><<<g, 128, 0, stream>>>(
        attn, (const float*)nullptr, posT, (const float*)nullptr, (const float*)nullptr, match,
        (int)HW, (int)D, (int)HW,
        HW * HW, 0, HW * D, 0, 1.0f);
  }
}
